// BidirectionalCrossAttention_79405355368740
// MI455X (gfx1250) — matrix operand order repaired and checked by reading, not compiled, not run on hardware
//
#include <hip/hip_runtime.h>
#include <hip/hip_bf16.h>

// Problem constants (fixed by the reference)
#define B_     2
#define SEQ    2048          // I == J
#define DIM_   1024
#define H_     16
#define DH_    64
#define INNER_ 1024
#define SCALE_ 0.125f        // 64^-0.5

typedef __attribute__((ext_vector_type(16))) _Float16 v16h;
typedef __attribute__((ext_vector_type(8)))  _Float16 v8h;
typedef __attribute__((ext_vector_type(8)))  float    v8f;
typedef __attribute__((ext_vector_type(4)))  unsigned int v4u;
typedef __attribute__((ext_vector_type(8)))  int v8i;
typedef __attribute__((ext_vector_type(4)))  int v4i;

// ---------------------------------------------------------------------------
// f32 -> f16 conversion (grid-stride)
// ---------------------------------------------------------------------------
__global__ void cvt_f32_to_f16(const float* __restrict__ in,
                               _Float16* __restrict__ out, long n) {
  long i = (long)blockIdx.x * blockDim.x + threadIdx.x;
  long stride = (long)gridDim.x * blockDim.x;
  for (; i < n; i += stride) out[i] = (_Float16)in[i];
}

// ---------------------------------------------------------------------------
// Transposing f32 -> f16 conversion: WT[n,k] = (f16)W[k,n]. 32x32 LDS tiles.
// ---------------------------------------------------------------------------
__global__ __launch_bounds__(256) void transpose_cvt_f16(
    const float* __restrict__ W, _Float16* __restrict__ WT, int K, int N) {
  __shared__ _Float16 t[32][33];
  const int kb = blockIdx.y * 32, nb = blockIdx.x * 32;
  const int tx = threadIdx.x & 31, ty = threadIdx.x >> 5;  // 32 x 8
#pragma unroll
  for (int r = 0; r < 32; r += 8)
    t[ty + r][tx] = (_Float16)W[(long)(kb + ty + r) * N + nb + tx];
  __syncthreads();
#pragma unroll
  for (int r = 0; r < 32; r += 8)
    WT[(long)(nb + ty + r) * K + kb + tx] = t[tx][ty + r];
}

// ---------------------------------------------------------------------------
// Tensor Data Mover: stage a 2D f16 tile (rowCount x rowLen elems) from
// global memory into LDS. Hand-packed D# per CDNA5 ISA (group0 + group1,
// 2D tile, data_size = 2 bytes, no gather/iterate/pad/multicast).
// This toolchain exposes the 6-arg builtin:
//   (uint32x4 g0, int32x8 g1, int32x4 g2, int32x4 g3, int32x8 g4, i32 cpol)
// ---------------------------------------------------------------------------
#if __has_builtin(__builtin_amdgcn_tensor_load_to_lds)
#define HAVE_TDM 1
__device__ __forceinline__ void tdm_load_tile_f16(
    const _Float16* gsrc, unsigned ldsOff, unsigned rowLen, unsigned rowCount,
    unsigned long long strideElems, unsigned long long tensorDim0,
    unsigned tensorDim1) {
  unsigned long long ga = (unsigned long long)(const void*)gsrc;
  v4u g0;
  g0[0] = 1u;                                           // count=1, user D#
  g0[1] = ldsOff;                                       // lds_addr (bytes)
  g0[2] = (unsigned)(ga & 0xFFFFFFFFu);                 // global_addr[31:0]
  g0[3] = (unsigned)((ga >> 32) & 0x01FFFFFFu) | (2u << 30);  // [56:32]|type=2
  v8i g1;
  g1[0] = (int)(1u << 16);                              // data_size=1 (2B)
  g1[1] = (int)((tensorDim0 & 0xFFFFu) << 16);          // tensor_dim0[15:0]
  g1[2] = (int)(((tensorDim0 >> 16) & 0xFFFFu) |
                ((tensorDim1 & 0xFFFFu) << 16));        // dim0 hi | dim1 lo
  g1[3] = (int)(((tensorDim1 >> 16) & 0xFFFFu) |
                ((rowLen & 0xFFFFu) << 16));            // dim1 hi | tile_dim0
  g1[4] = (int)(rowCount & 0xFFFFu);                    // tile_dim1, tile_dim2=0
  g1[5] = (int)(strideElems & 0xFFFFFFFFu);             // dim0_stride[31:0]
  g1[6] = (int)((strideElems >> 32) & 0xFFFFu);         // dim0_stride[47:32]
  g1[7] = 0;
  v4i z4 = {0, 0, 0, 0};
  v8i z8 = {0, 0, 0, 0, 0, 0, 0, 0};
  __builtin_amdgcn_tensor_load_to_lds(g0, g1, z4, z4, z8, 0);
}
#else
#define HAVE_TDM 0
#endif

// ---------------------------------------------------------------------------
// Universal NT WMMA GEMM: C[M,N] = A[M,K] x BT[N,K]^T  (both row-major).
// 256 threads = 8 waves; block tile 128(M) x 64(N); each wave: 16x64 strip,
// 4 accumulators sharing one A fragment -> 4 v_wmma per 32-K step.
// A block tile (128x32 f16) is staged in LDS by the TDM (wave 0 issues
// tensor_load_to_lds + s_wait_tensorcnt); BT fragments are contiguous b128
// global loads (lane = one BT row, K contiguous).
// Epilogue modes:
//   0: split-heads f16      [b,h,seq,dh]   (qk/cqk projections; M = B*SEQ)
//   1: split-heads-T f16    [b,h,dh,seq]   (v/cv projections, pre-transposed)
//   2: merge-heads f16      [b,seq,h*dh]   (AV products; z = b*H + h)
//   3: f32 + bias to d_out at dstBase      (output projections)
//   4: batched f16          [z,seq,seq]    (similarity scores; z = b*H + h)
// ---------------------------------------------------------------------------
__global__ __launch_bounds__(256) void gemm_nt_wmma(
    const _Float16* __restrict__ A, const _Float16* __restrict__ BT,
    void* __restrict__ Dst, const float* __restrict__ bias,
    int M, int N, int K, long aBatch, long bBatch,
    int mode, long dstBase, float scale) {
  __shared__ _Float16 tileA[128 * 32];

  const int lane  = threadIdx.x & 31;
  const int wave  = threadIdx.x >> 5;
  const int z     = blockIdx.z;
  const int nbase = blockIdx.x * 64;
  const int mbase = blockIdx.y * 128;
  const int r15   = lane & 15;
  const int koffA = (lane < 16) ? 0 : 8;    // 16-bit A lane layout
  const int koffB = (lane < 16) ? 0 : 8;    // 16-bit B lane layout

  const _Float16* Ab = A  + (long)z * aBatch;
  const _Float16* Bb = BT + (long)z * bBatch;

  const _Float16* brow[4];
#pragma unroll
  for (int nt = 0; nt < 4; ++nt)
    brow[nt] = Bb + (long)(nbase + nt * 16 + r15) * K;

  v8f acc[4] = {};

  for (int k0 = 0; k0 < K; k0 += 32) {
    __syncthreads();                       // previous tile fully consumed
#if HAVE_TDM
    if (wave == 0) {
      tdm_load_tile_f16(Ab + (long)mbase * K + k0,
                        (unsigned)(unsigned long long)(void*)&tileA[0],
                        /*rowLen=*/32, /*rowCount=*/128,
                        /*strideElems=*/(unsigned long long)K,
                        /*tensorDim0=*/(unsigned long long)K,
                        /*tensorDim1=*/(unsigned)M);
      __builtin_amdgcn_s_wait_tensorcnt(0);
    }
#else
    {
      int row = threadIdx.x >> 1, half = (threadIdx.x & 1) * 16;
      const _Float16* src = Ab + (long)(mbase + row) * K + k0 + half;
      *(v8h*)&tileA[row * 32 + half]     = *(const v8h*)src;
      *(v8h*)&tileA[row * 32 + half + 8] = *(const v8h*)(src + 8);
    }
#endif
    __syncthreads();                       // tile ready

    v8h a0 = *(const v8h*)&tileA[(wave * 16 + r15) * 32 + koffA];
    v8h a1 = *(const v8h*)&tileA[(wave * 16 + r15) * 32 + 16 + koffA];
    v16h a;
#pragma unroll
    for (int e = 0; e < 8; ++e) { a[e] = a0[e]; a[e + 8] = a1[e]; }

#pragma unroll
    for (int nt = 0; nt < 4; ++nt) {
      v8h b0 = *(const v8h*)(brow[nt] + k0 + koffB);
      v8h b1 = *(const v8h*)(brow[nt] + k0 + koffB + 16);
      v16h b;
#pragma unroll
      for (int e = 0; e < 8; ++e) { b[e] = b0[e]; b[e + 8] = b1[e]; }
      acc[nt] = __builtin_amdgcn_wmma_f32_16x16x32_f16(
          false, a, false, b, (short)0, acc[nt], false, false);
    }
    if (k0 + 32 < K)
      __builtin_prefetch(brow[0] + k0 + 64, 0, 0);   // global_prefetch_b8
  }

  const int mrow = mbase + wave * 16;
#pragma unroll
  for (int nt = 0; nt < 4; ++nt) {
    const int col = nbase + nt * 16 + r15;
#pragma unroll
    for (int r = 0; r < 8; ++r) {
      const int row = mrow + r + ((lane < 16) ? 0 : 8);
      const float v = acc[nt][r] * scale;
      if (mode == 0) {
        long b = row >> 11, i = row & 2047, h = col >> 6, d = col & 63;
        ((_Float16*)Dst)[((b * H_ + h) * (long)SEQ + i) * DH_ + d] =
            (_Float16)v;
      } else if (mode == 1) {
        long b = row >> 11, i = row & 2047, h = col >> 6, d = col & 63;
        ((_Float16*)Dst)[((b * H_ + h) * (long)DH_ + d) * SEQ + i] =
            (_Float16)v;
      } else if (mode == 2) {
        long b = z >> 4, h = z & 15;
        ((_Float16*)Dst)[b * (long)SEQ * INNER_ + (long)row * INNER_ +
                         h * DH_ + col] = (_Float16)v;
      } else if (mode == 3) {
        ((float*)Dst)[dstBase + (long)row * N + col] = v + bias[col];
      } else {
        ((_Float16*)Dst)[(long)z * SEQ * SEQ + (long)row * SEQ + col] =
            (_Float16)v;
      }
    }
  }
}

// ---------------------------------------------------------------------------
// Softmax statistics. One 256-thread block per row (or column) of 2048.
// ---------------------------------------------------------------------------
__global__ __launch_bounds__(256) void row_stats(
    const _Float16* __restrict__ Sim, float* __restrict__ rmax,
    float* __restrict__ rsum) {
  __shared__ float sm[8];
  long r = blockIdx.x;                         // flat (b*H + h)*SEQ + i
  const _Float16* p = Sim + r * (long)SEQ + threadIdx.x * 8;
  v8h d = *(const v8h*)p;
  float f[8];
#pragma unroll
  for (int e = 0; e < 8; ++e) f[e] = (float)d[e];

  float mx = f[0];
#pragma unroll
  for (int e = 1; e < 8; ++e) mx = fmaxf(mx, f[e]);
#pragma unroll
  for (int m = 16; m >= 1; m >>= 1) mx = fmaxf(mx, __shfl_xor(mx, m, 32));
  if ((threadIdx.x & 31) == 0) sm[threadIdx.x >> 5] = mx;
  __syncthreads();
  float gmx = sm[0];
#pragma unroll
  for (int w = 1; w < 8; ++w) gmx = fmaxf(gmx, sm[w]);
  __syncthreads();

  float s = 0.f;
#pragma unroll
  for (int e = 0; e < 8; ++e) s += __expf(f[e] - gmx);
#pragma unroll
  for (int m = 16; m >= 1; m >>= 1) s += __shfl_xor(s, m, 32);
  if ((threadIdx.x & 31) == 0) sm[threadIdx.x >> 5] = s;
  __syncthreads();
  if (threadIdx.x == 0) {
    float gs = 0.f;
    for (int w = 0; w < 8; ++w) gs += sm[w];
    rmax[r] = gmx;
    rsum[r] = gs;
  }
}

__global__ __launch_bounds__(256) void col_stats(
    const _Float16* __restrict__ Sim, float* __restrict__ cmax,
    float* __restrict__ csum) {
  __shared__ float sm[8];
  long cidx = blockIdx.x;                      // flat (b*H + h)*SEQ + j
  long z = cidx >> 11;
  int  j = (int)(cidx & 2047);
  const _Float16* p = Sim + z * (long)SEQ * SEQ + j;

  float f[8];
#pragma unroll
  for (int q = 0; q < 8; ++q)
    f[q] = (float)p[(long)(threadIdx.x + q * 256) * SEQ];

  float mx = f[0];
#pragma unroll
  for (int q = 1; q < 8; ++q) mx = fmaxf(mx, f[q]);
#pragma unroll
  for (int m = 16; m >= 1; m >>= 1) mx = fmaxf(mx, __shfl_xor(mx, m, 32));
  if ((threadIdx.x & 31) == 0) sm[threadIdx.x >> 5] = mx;
  __syncthreads();
  float gmx = sm[0];
#pragma unroll
  for (int w = 1; w < 8; ++w) gmx = fmaxf(gmx, sm[w]);
  __syncthreads();

  float s = 0.f;
#pragma unroll
  for (int q = 0; q < 8; ++q) s += __expf(f[q] - gmx);
#pragma unroll
  for (int m = 16; m >= 1; m >>= 1) s += __shfl_xor(s, m, 32);
  if ((threadIdx.x & 31) == 0) sm[threadIdx.x >> 5] = s;
  __syncthreads();
  if (threadIdx.x == 0) {
    float gs = 0.f;
    for (int w = 0; w < 8; ++w) gs += sm[w];
    cmax[cidx] = gmx;
    csum[cidx] = gs;
  }
}

// ---------------------------------------------------------------------------
// Fused dual softmax + talking-heads mix. One block per (b, 16x16 (i,j) tile);
// thread (ii,jj) owns position (i,j) across all heads. Overwrites Sim in
// place with the row-softmax mixed attention; writes the column-softmax mixed
// attention TRANSPOSED ([b,g,j,i]) so the second AV product is NT row-major.
// ---------------------------------------------------------------------------
__global__ __launch_bounds__(256) void softmax_talking_heads(
    _Float16* __restrict__ SimAttn, _Float16* __restrict__ CAttnT,
    const float* __restrict__ thw, const float* __restrict__ cthw,
    const float* __restrict__ rmax, const float* __restrict__ rsum,
    const float* __restrict__ cmax, const float* __restrict__ csum) {
  __shared__ float sThw[256], sCthw[256];
  __shared__ float sRm[16][16], sRs[16][16], sCm[16][16], sCs[16][16];

  const int b  = blockIdx.z;
  const int ib = blockIdx.y * 16, jb = blockIdx.x * 16;
  const int t  = threadIdx.x;
  const int jj = t & 15, ii = t >> 4;

  sThw[t]  = thw[t];
  sCthw[t] = cthw[t];
  {
    int h = t >> 4, q = t & 15;
    long base = (long)(b * H_ + h) * SEQ;
    sRm[h][q] = rmax[base + ib + q];
    sRs[h][q] = 1.0f / rsum[base + ib + q];
    sCm[h][q] = cmax[base + jb + q];
    sCs[h][q] = 1.0f / csum[base + jb + q];
  }
  __syncthreads();

  const long i = ib + ii, j = jb + jj;
  float pr[16], pc[16];
#pragma unroll
  for (int h = 0; h < 16; ++h) {
    float s = (float)SimAttn[((long)(b * H_ + h) * SEQ + i) * SEQ + j];
    pr[h] = __expf(s - sRm[h][ii]) * sRs[h][ii];
    pc[h] = __expf(s - sCm[h][jj]) * sCs[h][jj];
  }
#pragma unroll
  for (int g = 0; g < 16; ++g) {
    float ar = 0.f, ac = 0.f;
#pragma unroll
    for (int h = 0; h < 16; ++h) {
      ar = fmaf(sThw[g * 16 + h], pr[h], ar);
      ac = fmaf(sCthw[g * 16 + h], pc[h], ac);
    }
    SimAttn[((long)(b * H_ + g) * SEQ + i) * SEQ + j] = (_Float16)ar;
    CAttnT [((long)(b * H_ + g) * SEQ + j) * SEQ + i] = (_Float16)ac;
  }
}

// ---------------------------------------------------------------------------
// Host-side orchestration
// ---------------------------------------------------------------------------
extern "C" void kernel_launch(void* const* d_in, const int* in_sizes, int n_in,
                              void* d_out, int out_size, void* d_ws,
                              size_t ws_size, hipStream_t stream) {
  const float* x     = (const float*)d_in[0];
  const float* ctx   = (const float*)d_in[1];
  const float* Wqk   = (const float*)d_in[2];
  const float* Wcqk  = (const float*)d_in[3];
  const float* Wv    = (const float*)d_in[4];
  const float* Wcv   = (const float*)d_in[5];
  const float* Wout  = (const float*)d_in[6];
  const float* bout  = (const float*)d_in[7];
  const float* Wcout = (const float*)d_in[8];
  const float* bcout = (const float*)d_in[9];
  const float* thw   = (const float*)d_in[10];
  const float* cthw  = (const float*)d_in[11];

  char* ws = (char*)d_ws;
  size_t off = 0;
  auto alloc = [&](size_t bytes) -> void* {
    void* p = ws + off;
    off = (off + bytes + 255) & ~(size_t)255;
    return p;
  };

  const size_t nAct  = (size_t)B_ * SEQ * DIM_;     // 4M elems
  const size_t nW    = (size_t)DIM_ * INNER_;       // 1M elems
  const size_t nHead = (size_t)B_ * H_ * SEQ * DH_; // 4M elems
  const size_t nSim  = (size_t)B_ * H_ * SEQ * SEQ; // 128M elems

  _Float16* xF     = (_Float16*)alloc(nAct * 2);
  _Float16* cF     = (_Float16*)alloc(nAct * 2);
  _Float16* wqkT   = (_Float16*)alloc(nW * 2);      // transposed f16 weights
  _Float16* wcqkT  = (_Float16*)alloc(nW * 2);
  _Float16* wvT    = (_Float16*)alloc(nW * 2);
  _Float16* wcvT   = (_Float16*)alloc(nW * 2);
  _Float16* woutT  = (_Float16*)alloc(nW * 2);
  _Float16* wcoutT = (_Float16*)alloc(nW * 2);
  _Float16* qkF    = (_Float16*)alloc(nHead * 2);   // [b,h,seq,d]
  _Float16* cqkF   = (_Float16*)alloc(nHead * 2);   // [b,h,seq,d]
  _Float16* vT     = (_Float16*)alloc(nHead * 2);   // [b,h,d,seq]
  _Float16* cvT    = (_Float16*)alloc(nHead * 2);   // [b,h,d,seq]
  _Float16* simF   = (_Float16*)alloc(nSim * 2);    // reused as mixed attn
  _Float16* catT   = (_Float16*)alloc(nSim * 2);    // mixed cattn, transposed
  float* rmax = (float*)alloc((size_t)B_ * H_ * SEQ * 4);
  float* rsum = (float*)alloc((size_t)B_ * H_ * SEQ * 4);
  float* cmax = (float*)alloc((size_t)B_ * H_ * SEQ * 4);
  float* csum = (float*)alloc((size_t)B_ * H_ * SEQ * 4);
  _Float16* outM = (_Float16*)alloc((size_t)B_ * SEQ * INNER_ * 2);
  _Float16* ctxM = (_Float16*)alloc((size_t)B_ * SEQ * INNER_ * 2);

  // 1) activation conversions + transposed weight conversions
  cvt_f32_to_f16<<<1024, 256, 0, stream>>>(x, xF, (long)nAct);
  cvt_f32_to_f16<<<1024, 256, 0, stream>>>(ctx, cF, (long)nAct);
  dim3 gT(INNER_ / 32, DIM_ / 32, 1);
  transpose_cvt_f16<<<gT, 256, 0, stream>>>(Wqk,   wqkT,   DIM_, INNER_);
  transpose_cvt_f16<<<gT, 256, 0, stream>>>(Wcqk,  wcqkT,  DIM_, INNER_);
  transpose_cvt_f16<<<gT, 256, 0, stream>>>(Wv,    wvT,    DIM_, INNER_);
  transpose_cvt_f16<<<gT, 256, 0, stream>>>(Wcv,   wcvT,   DIM_, INNER_);
  transpose_cvt_f16<<<gT, 256, 0, stream>>>(Wout,  woutT,  INNER_, DIM_);
  transpose_cvt_f16<<<gT, 256, 0, stream>>>(Wcout, wcoutT, INNER_, DIM_);

  // 2) input projections (NT): qk/cqk split-heads, v/cv split-heads-T
  dim3 gProj(INNER_ / 64, (B_ * SEQ) / 128, 1);
  gemm_nt_wmma<<<gProj, 256, 0, stream>>>(xF, wqkT, qkF, nullptr,
      B_ * SEQ, INNER_, DIM_, 0, 0, 0, 0, 1.0f);
  gemm_nt_wmma<<<gProj, 256, 0, stream>>>(cF, wcqkT, cqkF, nullptr,
      B_ * SEQ, INNER_, DIM_, 0, 0, 0, 0, 1.0f);
  gemm_nt_wmma<<<gProj, 256, 0, stream>>>(xF, wvT, vT, nullptr,
      B_ * SEQ, INNER_, DIM_, 0, 0, 1, 0, 1.0f);
  gemm_nt_wmma<<<gProj, 256, 0, stream>>>(cF, wcvT, cvT, nullptr,
      B_ * SEQ, INNER_, DIM_, 0, 0, 1, 0, 1.0f);

  // 3) sim = scale * qk . cqk^T  (NT, batched over z = b*H + h)
  dim3 gSim(SEQ / 64, SEQ / 128, B_ * H_);
  gemm_nt_wmma<<<gSim, 256, 0, stream>>>(qkF, cqkF, simF, nullptr,
      SEQ, SEQ, DH_, (long)SEQ * DH_, (long)SEQ * DH_, 4, 0, SCALE_);

  // 4) softmax statistics along j (rows) and i (columns)
  row_stats<<<B_ * H_ * SEQ, 256, 0, stream>>>(simF, rmax, rsum);
  col_stats<<<B_ * H_ * SEQ, 256, 0, stream>>>(simF, cmax, csum);

  // 5) fused dual softmax + talking-heads (in-place attn; transposed cattn)
  dim3 gMix(SEQ / 16, SEQ / 16, B_);
  softmax_talking_heads<<<gMix, 256, 0, stream>>>(simF, catT, thw, cthw,
                                                  rmax, rsum, cmax, csum);

  // 6) AV products (NT: B = pre-transposed v/cv) -> merged heads f16
  dim3 gAV(DH_ / 64, SEQ / 128, B_ * H_);
  gemm_nt_wmma<<<gAV, 256, 0, stream>>>(simF, cvT, outM, nullptr,
      SEQ, DH_, SEQ, (long)SEQ * SEQ, (long)DH_ * SEQ, 2, 0, 1.0f);
  gemm_nt_wmma<<<gAV, 256, 0, stream>>>(catT, vT, ctxM, nullptr,
      SEQ, DH_, SEQ, (long)SEQ * SEQ, (long)DH_ * SEQ, 2, 0, 1.0f);

  // 7) output projections + bias -> f32 d_out (out, then context_out)
  dim3 gOut(DIM_ / 64, (B_ * SEQ) / 128, 1);
  gemm_nt_wmma<<<gOut, 256, 0, stream>>>(outM, woutT, d_out, bout,
      B_ * SEQ, DIM_, INNER_, 0, 0, 3, 0, 1.0f);
  gemm_nt_wmma<<<gOut, 256, 0, stream>>>(ctxM, wcoutT, d_out, bcout,
      B_ * SEQ, DIM_, INNER_, 0, 0, 3, (long)B_ * SEQ * DIM_, 1.0f);
}
